// DeformableTransformerBackbone_40853728920106
// MI455X (gfx1250) — matrix operand order repaired and checked by reading, not compiled, not run on hardware
//
#include <hip/hip_runtime.h>
#include <math.h>

#define NLVL   3
#define NHD    8
#define NPT    4
#define HDIM   32
#define DMODEL 256
#define DFF    512
#define LTOT   12096
#define NB     2
#define BTOT   (NB * LTOT)

typedef __attribute__((ext_vector_type(16))) _Float16 v16h;
typedef __attribute__((ext_vector_type(8)))  float    v8f;
typedef __attribute__((ext_vector_type(4)))  int      v4i_;
typedef __attribute__((address_space(1))) v4i_* as1_v4i;
typedef __attribute__((address_space(3))) v4i_* as3_v4i;

union AFrag { unsigned int u[8]; v16h h; };
union H4 { _Float16 h[4]; unsigned long long u64; };

#if defined(__has_builtin)
#  if __has_builtin(__builtin_amdgcn_global_load_async_to_lds_b128)
#    define ASYNC_B_STAGE 1
#  endif
#endif
#ifndef ASYNC_B_STAGE
#  define ASYNC_B_STAGE 0
#endif

// ---------------------------------------------------------------------------
// Sine positional embedding + level embedding + reference points
// ---------------------------------------------------------------------------
__global__ void pos_ref_kernel(const float* __restrict__ level_embed,
                               float* __restrict__ pos, float* __restrict__ ref) {
  int idx = blockIdx.x * blockDim.x + threadIdx.x;
  if (idx >= LTOT * DMODEL) return;
  int t = idx >> 8;
  int d = idx & 255;
  int l, H, W, off;
  if (t < 9216)       { l = 0; H = 96; W = 96; off = 0; }
  else if (t < 11520) { l = 1; H = 48; W = 48; off = 9216; }
  else                { l = 2; H = 24; W = 24; off = 11520; }
  int p = t - off;
  int h = p / W, w = p % W;
  const float scale = 6.283185307179586f;
  const float eps = 1e-6f;
  bool isY = d < 128;
  int dd = isY ? d : d - 128;
  float coord = isY ? ((float)(h + 1) / ((float)H + eps)) * scale
                    : ((float)(w + 1) / ((float)W + eps)) * scale;
  int k = dd >> 1;
  float freq = powf(10000.0f, (2.0f * (float)k) / 128.0f);
  float v = coord / freq;
  float pe = (dd & 1) ? cosf(v) : sinf(v);
  pos[idx] = pe + level_embed[l * DMODEL + d];
  if (d == 0) ref[t * 2 + 0] = ((float)w + 0.5f) / (float)W;
  if (d == 1) ref[t * 2 + 1] = ((float)h + 0.5f) / (float)H;
}

// ---------------------------------------------------------------------------
// [C, HW] -> [HW, C] transpose (one batch image of one level)
// ---------------------------------------------------------------------------
__global__ void transpose_feat_kernel(const float* __restrict__ feat,
                                      float* __restrict__ ft, int C, int HW) {
  int idx = blockIdx.x * blockDim.x + threadIdx.x;
  if (idx >= C * HW) return;
  int hw = idx / C, c = idx % C;
  ft[idx] = feat[(size_t)c * HW + hw];
}

// ---------------------------------------------------------------------------
// Weight pre-pass: W[K][N] f32 -> Wt[N][K] f16 (transposed + converted once)
// ---------------------------------------------------------------------------
__global__ void wt_convert_kernel(const float* __restrict__ W,
                                  _Float16* __restrict__ out, int K, int N) {
  int idx = blockIdx.x * blockDim.x + threadIdx.x;
  if (idx >= K * N) return;
  int n = idx / K, k = idx % K;
  out[idx] = (_Float16)W[(size_t)k * N + n];
}

// ---------------------------------------------------------------------------
// Generic WMMA GEMM: Y[M,N] = A[M,K] (f32) x Wt[N][K] (f16) + bias (+res)(+relu)
// A staged with batched float4 loads (full MLP) + packed f16 ds_store_b64;
// B staged via GLOBAL_LOAD_ASYNC_TO_LDS_B128 when available (ASYNCcnt),
// else direct 16-byte f16 copies (pre-transposed weights).
// Block = 256 threads (8 wave32), block tile 128x64, K-tile 64 (2 WMMA steps
// per barrier), wave tile 32x32 (2x2 v_wmma_f32_16x16x32_f16).
// Requires: K % 64 == 0 (true for all uses here).
// ---------------------------------------------------------------------------
template <bool RELU, bool RESID>
__global__ __launch_bounds__(256) void gemm_wmma_kernel(
    const float* __restrict__ A, const _Float16* __restrict__ Wt,
    const float* __restrict__ bias, const float* __restrict__ resid,
    float* __restrict__ Y, int M, int N, int K) {
  __shared__ _Float16 sA[128 * 72];  // 128 rows x 64 K (stride 72 halves)
  __shared__ _Float16 sB[64 * 72];   // sB[n][k], 64 n x 64 k
  const int tid  = threadIdx.x;
  const int lane = tid & 31;
  const int wave = tid >> 5;
  const int wr = wave >> 1;   // 0..3 (M direction)
  const int wc = wave & 1;    // 0..1 (N direction)
  const int lh  = lane & 15;
  const int lhi = lane >> 4;
  const int m0 = blockIdx.y * 128;
  const int n0 = blockIdx.x * 64;

  v8f acc[2][2] = {};

  const int ktiles = K >> 6;
  for (int kt = 0; kt < ktiles; ++kt) {
    const int kbase = kt << 6;
    // ---- stage B: async DMA to LDS (preferred) or direct 16B copies ----
#if ASYNC_B_STAGE
#pragma unroll
    for (int i = 0; i < 2; ++i) {
      int f = tid + i * 256;          // 16B chunk id, 512 total (64 n x 8)
      int n = f >> 3;
      int c8 = (f & 7) << 3;
      int gn = n0 + n;
      gn = gn < N ? gn : N - 1;       // clamp: garbage cols masked at store
      __builtin_amdgcn_global_load_async_to_lds_b128(
          (as1_v4i)(v4i_*)(Wt + (size_t)gn * K + kbase + c8),
          (as3_v4i)(v4i_*)(&sB[n * 72 + c8]),
          0, 0);
    }
#else
    uint4 vb[2];
#pragma unroll
    for (int i = 0; i < 2; ++i) {
      int f = tid + i * 256;
      int n = f >> 3;
      int c8 = (f & 7) << 3;
      int gn = n0 + n;
      gn = gn < N ? gn : N - 1;
      vb[i] = *reinterpret_cast<const uint4*>(Wt + (size_t)gn * K + kbase + c8);
    }
#endif
    // ---- stage A: batch all 8 global_load_b128 first (MLP), then cvt+store
    float4 va[8];
#pragma unroll
    for (int i = 0; i < 8; ++i) {
      int f = tid + i * 256;          // float4 id, 2048 total
      int r = f >> 4;
      int c4 = (f & 15) << 2;
      int gm = m0 + r;
      gm = gm < M ? gm : M - 1;       // clamp: garbage rows masked at store
      va[i] = *reinterpret_cast<const float4*>(A + (size_t)gm * K + kbase + c4);
    }
#pragma unroll
    for (int i = 0; i < 8; ++i) {
      int f = tid + i * 256;
      int r = f >> 4;
      int c4 = (f & 15) << 2;
      H4 t;
      t.h[0] = (_Float16)va[i].x; t.h[1] = (_Float16)va[i].y;
      t.h[2] = (_Float16)va[i].z; t.h[3] = (_Float16)va[i].w;
      *reinterpret_cast<unsigned long long*>(&sA[r * 72 + c4]) = t.u64;
    }
#if !ASYNC_B_STAGE
#pragma unroll
    for (int i = 0; i < 2; ++i) {
      int f = tid + i * 256;
      int n = f >> 3;
      int c8 = (f & 7) << 3;
      *reinterpret_cast<uint4*>(&sB[n * 72 + c8]) = vb[i];
    }
#endif
    // speculative prefetch of next A K-slab (global_prefetch_b8)
    if (kt + 1 < ktiles) {
      int gm = m0 + (tid >> 1);
      gm = gm < M ? gm : M - 1;
      __builtin_prefetch(A + (size_t)gm * K + kbase + 64, 0, 0);
    }
#if ASYNC_B_STAGE
#  if defined(__has_builtin) && __has_builtin(__builtin_amdgcn_s_wait_asynccnt)
    __builtin_amdgcn_s_wait_asynccnt(0);
#  else
    asm volatile("s_wait_asynccnt 0x0" ::: "memory");
#  endif
#endif
    __syncthreads();

    // ---- 2 WMMA K-steps per staged tile ----
#pragma unroll
    for (int ks = 0; ks < 2; ++ks) {
      const int ko = ks * 32;
      AFrag af[2], bf[2];
#pragma unroll
      for (int mi = 0; mi < 2; ++mi) {
        const _Float16* base = &sA[(wr * 32 + mi * 16 + lh) * 72 + ko];
        int kofs = lhi * 16;
#pragma unroll
        for (int v = 0; v < 4; ++v)
          af[mi].u[v] = *(const unsigned int*)(base + kofs + 2 * v);
#pragma unroll
        for (int v = 0; v < 4; ++v)
          af[mi].u[4 + v] = *(const unsigned int*)(base + 8 + kofs + 2 * v);
      }
#pragma unroll
      for (int ni = 0; ni < 2; ++ni) {
        const _Float16* base = &sB[(wc * 32 + ni * 16 + lh) * 72 + ko];
        int kofs = lhi * 16;
#pragma unroll
        for (int v = 0; v < 8; ++v)
          bf[ni].u[v] = *(const unsigned int*)(base + kofs + 2 * v);
      }
#pragma unroll
      for (int mi = 0; mi < 2; ++mi)
#pragma unroll
        for (int ni = 0; ni < 2; ++ni)
          acc[mi][ni] = __builtin_amdgcn_wmma_f32_16x16x32_f16(
              false, af[mi].h, false, bf[ni].h, (short)0, acc[mi][ni],
              false, false);
    }
    __syncthreads();
  }

  // epilogue: bias (+residual)(+relu); C/D layout: VGPR v -> row (lhi*8+v)
#pragma unroll
  for (int mi = 0; mi < 2; ++mi) {
#pragma unroll
    for (int ni = 0; ni < 2; ++ni) {
      int col = n0 + wc * 32 + ni * 16 + lh;
#pragma unroll
      for (int v = 0; v < 8; ++v) {
        int row = m0 + wr * 32 + mi * 16 + lhi * 8 + v;
        if (row < M && col < N) {
          float o = acc[mi][ni][v] + bias[col];
          if (RESID) o += resid[(size_t)row * N + col];
          if (RELU)  o = fmaxf(o, 0.0f);
          Y[(size_t)row * N + col] = o;
        }
      }
    }
  }
}

// ---------------------------------------------------------------------------
// GroupNorm (32 groups over D=256, reduce over 8 channels x HW per batch)
// ---------------------------------------------------------------------------
__global__ void gn_reduce_kernel(const float* __restrict__ s, float* __restrict__ stat,
                                 int HW, int start) {
  int b = blockIdx.x >> 5;
  int g = blockIdx.x & 31;
  const float* base = s + (size_t)(b * LTOT + start) * DMODEL;
  int n = HW * 8;
  float sum = 0.f, sq = 0.f;
  for (int i = threadIdx.x; i < n; i += 256) {
    int hw = i >> 3, ch = i & 7;
    float v = base[(size_t)hw * DMODEL + g * 8 + ch];
    sum += v; sq += v * v;
  }
  __shared__ float s1[256], s2[256];
  s1[threadIdx.x] = sum; s2[threadIdx.x] = sq;
  __syncthreads();
  for (int st2 = 128; st2 > 0; st2 >>= 1) {
    if (threadIdx.x < st2) {
      s1[threadIdx.x] += s1[threadIdx.x + st2];
      s2[threadIdx.x] += s2[threadIdx.x + st2];
    }
    __syncthreads();
  }
  if (threadIdx.x == 0) {
    float m = s1[0] / (float)n;
    stat[blockIdx.x * 2 + 0] = m;
    stat[blockIdx.x * 2 + 1] = s2[0] / (float)n - m * m;
  }
}

__global__ void gn_norm_kernel(float* __restrict__ s, const float* __restrict__ stat,
                               const float* __restrict__ gg, const float* __restrict__ gb,
                               int HW, int start) {
  int idx = blockIdx.x * blockDim.x + threadIdx.x;
  if (idx >= NB * HW * DMODEL) return;
  int d = idx & 255;
  int r = idx >> 8;
  int b = r / HW;
  int hw = r % HW;
  int g = d >> 3;
  float m = stat[(b * 32 + g) * 2 + 0];
  float var = stat[(b * 32 + g) * 2 + 1];
  size_t o = (size_t)(b * LTOT + start + hw) * DMODEL + d;
  float x = s[o];
  s[o] = (x - m) * rsqrtf(var + 1e-5f) * gg[d] + gb[d];
}

// ---------------------------------------------------------------------------
__global__ void add_pos_kernel(const float* __restrict__ src,
                               const float* __restrict__ pos,
                               float* __restrict__ q) {
  int idx = blockIdx.x * blockDim.x + threadIdx.x;
  if (idx >= BTOT * DMODEL) return;
  int row = idx >> 8;
  int t = row % LTOT;
  q[idx] = src[idx] + pos[t * DMODEL + (idx & 255)];
}

// softmax over 12 sampling weights per (b,t,head)
__global__ void softmax12_kernel(float* __restrict__ aw) {
  int r = blockIdx.x * blockDim.x + threadIdx.x;
  if (r >= BTOT * NHD) return;
  int token = r >> 3, h = r & 7;
  float* p = aw + (size_t)token * 96 + h * 12;
  float mx = -1e30f;
  for (int i = 0; i < 12; ++i) mx = fmaxf(mx, p[i]);
  float e[12], sm = 0.f;
  for (int i = 0; i < 12; ++i) { e[i] = expf(p[i] - mx); sm += e[i]; }
  float inv = 1.0f / sm;
  for (int i = 0; i < 12; ++i) p[i] = e[i] * inv;
}

// ---------------------------------------------------------------------------
// Deformable attention: one wave32 per (b, token, head); lane == head channel.
// ---------------------------------------------------------------------------
__global__ __launch_bounds__(256) void deform_attn_kernel(
    const float* __restrict__ val, const float* __restrict__ offs,
    const float* __restrict__ aw, const float* __restrict__ ref,
    float* __restrict__ out) {
  int wid = blockIdx.x * 8 + (threadIdx.x >> 5);
  if (wid >= BTOT * NHD) return;
  int lane = threadIdx.x & 31;
  int row = wid >> 3;  // b*LTOT + t
  int h = wid & 7;
  int t = row % LTOT;
  int b = row / LTOT;
  float rx = ref[t * 2 + 0], ry = ref[t * 2 + 1];
  const float* orow = offs + (size_t)row * 192 + h * 24;
  const float* arow = aw + (size_t)row * 96 + h * 12;
  const int Wl[3] = {96, 48, 24};
  const int Hl[3] = {96, 48, 24};
  const int st[3] = {0, 9216, 11520};
  float acc = 0.f;
#pragma unroll
  for (int l = 0; l < 3; ++l) {
    const int Wc = Wl[l], Hc = Hl[l];
    float gxb = rx * (float)Wc - 0.5f;
    float gyb = ry * (float)Hc - 0.5f;
#pragma unroll
    for (int p = 0; p < 4; ++p) {
      float gx = gxb + orow[l * 8 + p * 2 + 0];
      float gy = gyb + orow[l * 8 + p * 2 + 1];
      float a = arow[l * 4 + p];
      float x0f = floorf(gx), y0f = floorf(gy);
      float fx = gx - x0f, fy = gy - y0f;
      int x0 = (int)x0f, y0 = (int)y0f;
#pragma unroll
      for (int dy = 0; dy < 2; ++dy) {
#pragma unroll
        for (int dx = 0; dx < 2; ++dx) {
          int xi = x0 + dx, yi = y0 + dy;
          if (xi >= 0 && xi < Wc && yi >= 0 && yi < Hc) {
            float w = (dx ? fx : 1.f - fx) * (dy ? fy : 1.f - fy);
            size_t vrow = (size_t)b * LTOT + st[l] + yi * Wc + xi;
            acc += w * a * val[vrow * DMODEL + h * HDIM + lane];
          }
        }
      }
    }
  }
  out[(size_t)row * DMODEL + h * HDIM + lane] = acc;
}

// ---------------------------------------------------------------------------
// LayerNorm over D=256 (one block per token)
// ---------------------------------------------------------------------------
__global__ void layernorm_kernel(const float* __restrict__ x,
                                 const float* __restrict__ g,
                                 const float* __restrict__ bta,
                                 float* __restrict__ y) {
  int row = blockIdx.x;
  int d = threadIdx.x;
  float v = x[(size_t)row * DMODEL + d];
  __shared__ float s1[256], s2[256];
  s1[d] = v; s2[d] = v * v;
  __syncthreads();
  for (int st = 128; st > 0; st >>= 1) {
    if (d < st) { s1[d] += s1[d + st]; s2[d] += s2[d + st]; }
    __syncthreads();
  }
  float m = s1[0] / 256.0f;
  float var = s2[0] / 256.0f - m * m;
  y[(size_t)row * DMODEL + d] = (v - m) * rsqrtf(var + 1e-5f) * g[d] + bta[d];
}

// ---------------------------------------------------------------------------
extern "C" void kernel_launch(void* const* d_in, const int* in_sizes, int n_in,
                              void* d_out, int out_size, void* d_ws, size_t ws_size,
                              hipStream_t stream) {
  const float* feat[3] = {(const float*)d_in[0], (const float*)d_in[1], (const float*)d_in[2]};
  const float* pw[3]   = {(const float*)d_in[3], (const float*)d_in[4], (const float*)d_in[5]};
  const float* pb          = (const float*)d_in[6];
  const float* gn_g        = (const float*)d_in[7];
  const float* gn_b        = (const float*)d_in[8];
  const float* level_embed = (const float*)d_in[9];
  const float* so_w = (const float*)d_in[10];
  const float* so_b = (const float*)d_in[11];
  const float* aw_w = (const float*)d_in[12];
  const float* aw_b = (const float*)d_in[13];
  const float* vp_w = (const float*)d_in[14];
  const float* vp_b = (const float*)d_in[15];
  const float* op_w = (const float*)d_in[16];
  const float* op_b = (const float*)d_in[17];
  const float* ln1_g = (const float*)d_in[18];
  const float* ln1_b = (const float*)d_in[19];
  const float* l1_w  = (const float*)d_in[20];
  const float* l1_b  = (const float*)d_in[21];
  const float* l2_w  = (const float*)d_in[22];
  const float* l2_b  = (const float*)d_in[23];
  const float* ln2_g = (const float*)d_in[24];
  const float* ln2_b = (const float*)d_in[25];

  float* ws = (float*)d_ws;
  size_t off = 0;
  float* wSRC = ws + off; off += (size_t)BTOT * DMODEL;
  float* wPOS = ws + off; off += (size_t)LTOT * DMODEL;
  float* wREF = ws + off; off += (size_t)LTOT * 2;
  float* wQ   = ws + off; off += (size_t)BTOT * DMODEL;
  float* wOFF = ws + off; off += (size_t)BTOT * 192;
  float* wAW  = ws + off; off += (size_t)BTOT * 96;
  float* wVAL = ws + off; off += (size_t)BTOT * DMODEL;
  float* wATT = ws + off; off += (size_t)BTOT * DMODEL;
  float* wY   = ws + off; off += (size_t)BTOT * DMODEL;
  float* wH1  = ws + off; off += (size_t)BTOT * DFF;
  float* wFT  = wH1;            // alias: used only during input projection
  float* wGST = wH1 + 600000;   // alias: GN statistics (64*2 floats)

  // f16 pre-transposed weight arena (after all f32 buffers)
  _Float16* w16 = (_Float16*)(ws + off);
  size_t ho = 0;
  const int Hs[3] = {96, 48, 24}, Wsz[3] = {96, 48, 24};
  const int Cs[3] = {64, 128, 256}, st[3] = {0, 9216, 11520};
  _Float16* wPW16[3];
  for (int l = 0; l < 3; ++l) { wPW16[l] = w16 + ho; ho += (size_t)Cs[l] * DMODEL; }
  _Float16 *wSO16[3], *wAWW16[3], *wVP16[3], *wOP16[3], *wL116[3], *wL216[3];
  for (int i = 0; i < 3; ++i) {
    wSO16[i]  = w16 + ho; ho += (size_t)DMODEL * 192;
    wAWW16[i] = w16 + ho; ho += (size_t)DMODEL * 96;
    wVP16[i]  = w16 + ho; ho += (size_t)DMODEL * DMODEL;
    wOP16[i]  = w16 + ho; ho += (size_t)DMODEL * DMODEL;
    wL116[i]  = w16 + ho; ho += (size_t)DMODEL * DFF;
    wL216[i]  = w16 + ho; ho += (size_t)DFF * DMODEL;
  }

  // ---- weight pre-pass: convert + transpose all GEMM weights to f16 ----
  for (int l = 0; l < 3; ++l) {
    int tot = Cs[l] * DMODEL;
    wt_convert_kernel<<<(tot + 255) / 256, 256, 0, stream>>>(pw[l], wPW16[l],
                                                             Cs[l], DMODEL);
  }
  for (int i = 0; i < 3; ++i) {
    int t1 = DMODEL * 192;
    wt_convert_kernel<<<(t1 + 255) / 256, 256, 0, stream>>>(
        so_w + (size_t)i * DMODEL * 192, wSO16[i], DMODEL, 192);
    int t2 = DMODEL * 96;
    wt_convert_kernel<<<(t2 + 255) / 256, 256, 0, stream>>>(
        aw_w + (size_t)i * DMODEL * 96, wAWW16[i], DMODEL, 96);
    int t3 = DMODEL * DMODEL;
    wt_convert_kernel<<<(t3 + 255) / 256, 256, 0, stream>>>(
        vp_w + (size_t)i * DMODEL * DMODEL, wVP16[i], DMODEL, DMODEL);
    wt_convert_kernel<<<(t3 + 255) / 256, 256, 0, stream>>>(
        op_w + (size_t)i * DMODEL * DMODEL, wOP16[i], DMODEL, DMODEL);
    int t4 = DMODEL * DFF;
    wt_convert_kernel<<<(t4 + 255) / 256, 256, 0, stream>>>(
        l1_w + (size_t)i * DMODEL * DFF, wL116[i], DMODEL, DFF);
    wt_convert_kernel<<<(t4 + 255) / 256, 256, 0, stream>>>(
        l2_w + (size_t)i * DFF * DMODEL, wL216[i], DFF, DMODEL);
  }

  // positional embedding + reference points
  {
    int tot = LTOT * DMODEL;
    pos_ref_kernel<<<(tot + 255) / 256, 256, 0, stream>>>(level_embed, wPOS, wREF);
  }

  // input projection (WMMA GEMM) + GroupNorm per level
  for (int l = 0; l < 3; ++l) {
    int HW = Hs[l] * Wsz[l], C = Cs[l];
    for (int b = 0; b < NB; ++b) {
      int tot = HW * C;
      transpose_feat_kernel<<<(tot + 255) / 256, 256, 0, stream>>>(
          feat[l] + (size_t)b * C * HW, wFT, C, HW);
      dim3 grid((DMODEL + 63) / 64, (HW + 127) / 128);
      gemm_wmma_kernel<false, false><<<grid, 256, 0, stream>>>(
          wFT, wPW16[l], pb + l * DMODEL, nullptr,
          wSRC + (size_t)(b * LTOT + st[l]) * DMODEL, HW, DMODEL, C);
    }
    gn_reduce_kernel<<<NB * 32, 256, 0, stream>>>(wSRC, wGST, HW, st[l]);
    int tot = NB * HW * DMODEL;
    gn_norm_kernel<<<(tot + 255) / 256, 256, 0, stream>>>(
        wSRC, wGST, gn_g + l * DMODEL, gn_b + l * DMODEL, HW, st[l]);
  }

  // encoder layers
  dim3 g192((192 + 63) / 64, (BTOT + 127) / 128);
  dim3 g96 ((96  + 63) / 64, (BTOT + 127) / 128);
  dim3 g256((256 + 63) / 64, (BTOT + 127) / 128);
  dim3 g512((512 + 63) / 64, (BTOT + 127) / 128);
  for (int i = 0; i < 3; ++i) {
    int tot = BTOT * DMODEL;
    add_pos_kernel<<<(tot + 255) / 256, 256, 0, stream>>>(wSRC, wPOS, wQ);
    gemm_wmma_kernel<false, false><<<g192, 256, 0, stream>>>(
        wQ, wSO16[i], so_b + i * 192, nullptr, wOFF, BTOT, 192, DMODEL);
    gemm_wmma_kernel<false, false><<<g96, 256, 0, stream>>>(
        wQ, wAWW16[i], aw_b + i * 96, nullptr, wAW, BTOT, 96, DMODEL);
    softmax12_kernel<<<(BTOT * NHD + 255) / 256, 256, 0, stream>>>(wAW);
    gemm_wmma_kernel<false, false><<<g256, 256, 0, stream>>>(
        wSRC, wVP16[i], vp_b + i * DMODEL, nullptr, wVAL, BTOT, DMODEL, DMODEL);
    deform_attn_kernel<<<(BTOT * NHD + 7) / 8, 256, 0, stream>>>(
        wVAL, wOFF, wAW, wREF, wATT);
    gemm_wmma_kernel<false, true><<<g256, 256, 0, stream>>>(
        wATT, wOP16[i], op_b + i * DMODEL, wSRC, wY, BTOT, DMODEL, DMODEL);
    layernorm_kernel<<<BTOT, 256, 0, stream>>>(
        wY, ln1_g + i * DMODEL, ln1_b + i * DMODEL, wSRC);
    gemm_wmma_kernel<true, false><<<g512, 256, 0, stream>>>(
        wSRC, wL116[i], l1_b + i * DFF, nullptr, wH1, BTOT, DFF, DMODEL);
    gemm_wmma_kernel<false, true><<<g256, 256, 0, stream>>>(
        wH1, wL216[i], l2_b + i * DMODEL, wSRC, wY, BTOT, DMODEL, DFF);
    float* dst = (i == 2) ? (float*)d_out : wSRC;
    layernorm_kernel<<<BTOT, 256, 0, stream>>>(
        wY, ln2_g + i * DMODEL, ln2_b + i * DMODEL, dst);
  }
}
